// CtAslLoss_21869973471429
// MI455X (gfx1250) — matrix operand order repaired and checked by reading, not compiled, not run on hardware
//
#include <hip/hip_runtime.h>

constexpr int NB   = 64;    // batch
constexpr int NTOK = 196;   // tokens minus CLS
constexpr int ND   = 768;   // dim
constexpr int NLAY = 13;
constexpr int LYR  = 12;    // last layer
constexpr int SEQ  = 197;
constexpr int KSEL = 16;    // top-k
constexpr int CHUNKS = 7;   // row-chunks per batch in streaming kernel
constexpr int ROWS_PER_CHUNK = 28;  // 7*28 == 196

typedef float  v8f  __attribute__((ext_vector_type(8)));
typedef __bf16 bf16_t;
typedef bf16_t v16bf __attribute__((ext_vector_type(16)));
typedef bf16_t v8bf  __attribute__((ext_vector_type(8)));

static __device__ __forceinline__ bf16_t f2bf(float f) {
  unsigned u = __float_as_uint(f);
  u = (u + 0x7FFFu + ((u >> 16) & 1u)) >> 16;   // RNE truncate to bf16
  unsigned short s = (unsigned short)u;
  bf16_t h; __builtin_memcpy(&h, &s, 2); return h;
}
static __device__ __forceinline__ float bf2f(bf16_t h) {
  unsigned short s; __builtin_memcpy(&s, &h, 2);
  return __uint_as_float(((unsigned)s) << 16);
}
static __device__ __forceinline__ float wave_sum(float v) {
  for (int off = 16; off > 0; off >>= 1) v += __shfl_xor(v, off, 32);
  return v;
}

// ---------- K1: row softmax (beta) and L1-normalized labels (ny) ----------
__global__ __launch_bounds__(64) void k1_softmax(const float* __restrict__ labels,
                                                 float* __restrict__ beta,
                                                 float* __restrict__ ny) {
  const int b = threadIdx.x;
  if (b >= NB) return;
  float mx = -INFINITY;
  for (int m = 0; m < NB; ++m) mx = fmaxf(mx, labels[b * NB + m]);
  float se = 0.f, l1 = 0.f;
  for (int m = 0; m < NB; ++m) {
    float v = labels[b * NB + m];
    se += expf(v - mx); l1 += fabsf(v);
  }
  l1 = fmaxf(l1, 1e-12f);
  for (int m = 0; m < NB; ++m) {
    float v = labels[b * NB + m];
    beta[b * NB + m] = expf(v - mx) / se;
    ny[b * NB + m] = v / l1;
  }
}

// ---------- K2a: squared row norms of Lbert ----------
__global__ __launch_bounds__(256) void k2_nl(const float* __restrict__ bert,
                                             float* __restrict__ nL2) {
  const int m = blockIdx.x, tid = threadIdx.x;
  const float* row = bert + (size_t)LYR * NB * ND + (size_t)m * ND;
  float s = 0.f;
  for (int i = tid; i < ND; i += 256) { float v = row[i]; s += v * v; }
  s = wave_sum(s);
  __shared__ float sh[8];
  if ((tid & 31) == 0) sh[tid >> 5] = s;
  __syncthreads();
  if (tid == 0) { float t2 = 0.f; for (int w = 0; w < 8; ++w) t2 += sh[w]; nL2[m] = t2; }
}

// ---------- K2b: o = ny @ Lbert  (64x64)x(64x768) ----------
__global__ __launch_bounds__(256) void k2_o(const float* __restrict__ ny,
                                            const float* __restrict__ bert,
                                            float* __restrict__ o) {
  const int t = blockIdx.x * 256 + threadIdx.x;
  if (t >= NB * ND) return;
  const int b = t / ND, d = t % ND;
  const float* Lb = bert + (size_t)LYR * NB * ND;
  float acc = 0.f;
  for (int m = 0; m < NB; ++m) acc += ny[b * NB + m] * Lb[(size_t)m * ND + d];
  o[t] = acc;
}

// ---------- K2c: split Lbert into bf16 hi/lo for compensated WMMA ----------
__global__ __launch_bounds__(256) void k2_split(const float* __restrict__ bert,
                                                unsigned short* __restrict__ LbH,
                                                unsigned short* __restrict__ LbL) {
  const int t = blockIdx.x * 256 + threadIdx.x;
  if (t >= NB * ND) return;
  float v = bert[(size_t)LYR * NB * ND + t];
  bf16_t h = f2bf(v);
  bf16_t l = f2bf(v - bf2f(h));
  unsigned short hs, ls;
  __builtin_memcpy(&hs, &h, 2); __builtin_memcpy(&ls, &l, 2);
  LbH[t] = hs; LbL[t] = ls;
}

// ---------- K3: streaming pass over E (38.7MB) split 7-way per batch for BW ----------
__global__ __launch_bounds__(256) void k3_score(const float* __restrict__ vit,
                                                const float* __restrict__ o,
                                                float* __restrict__ score,
                                                float* __restrict__ nE2,
                                                float* __restrict__ fE2part) {
  __shared__ __align__(16) float os[ND];
  __shared__ float fes[8];
  const int b = blockIdx.x, ch = blockIdx.y;
  const int tid = threadIdx.x, lane = tid & 31, wid = tid >> 5;
  for (int i = tid; i < ND; i += 256) os[i] = o[b * ND + i];
  __syncthreads();
  const float* Eb = vit + ((size_t)(b * NLAY + LYR) * SEQ + 1) * ND;
  const int n0 = ch * ROWS_PER_CHUNK;
  float fe = 0.f;
  for (int nl = wid; nl < ROWS_PER_CHUNK; nl += 8) {
    const int n = n0 + nl;
    const float4* row = (const float4*)(Eb + (size_t)n * ND);
    if (nl + 8 < ROWS_PER_CHUNK)
      __builtin_prefetch(Eb + (size_t)(n + 8) * ND + lane * 4, 0, 0);
    float dot = 0.f, sq = 0.f;
#pragma unroll
    for (int j = 0; j < 6; ++j) {
      const int i4 = lane + j * 32;
      const float4 e = row[i4];
      const float4 ov = ((const float4*)os)[i4];
      dot += e.x * ov.x + e.y * ov.y + e.z * ov.z + e.w * ov.w;
      sq  += e.x * e.x + e.y * e.y + e.z * e.z + e.w * e.w;
    }
    dot = wave_sum(dot);
    sq  = wave_sum(sq);
    if (lane == 0) { score[b * NTOK + n] = dot; nE2[b * NTOK + n] = sq; fe += sq; }
  }
  if (lane == 0) fes[wid] = fe;
  __syncthreads();
  if (tid == 0) {
    float s = 0.f;
    for (int w = 0; w < 8; ++w) s += fes[w];
    fE2part[b * CHUNKS + ch] = s;
  }
}

// ---------- K3b: reduce per-chunk Frobenius partials ----------
__global__ __launch_bounds__(64) void k3b_reduce(const float* __restrict__ fE2part,
                                                 float* __restrict__ fE2) {
  const int b = threadIdx.x;
  if (b < NB) {
    float s = 0.f;
    for (int c = 0; c < CHUNKS; ++c) s += fE2part[b * CHUNKS + c];
    fE2[b] = s;
  }
}

// ---------- K4: top-16 per row (one wave per b) ----------
__global__ __launch_bounds__(32) void k4_topk(const float* __restrict__ score,
                                              int* __restrict__ topk) {
  const int b = blockIdx.x, lane = threadIdx.x;
  float vals[7];
#pragma unroll
  for (int j = 0; j < 7; ++j) {
    const int n = lane + j * 32;
    vals[j] = (n < NTOK) ? score[b * NTOK + n] : -INFINITY;
  }
  for (int k = 0; k < KSEL; ++k) {
    float best = -INFINITY; int bj = -1;
#pragma unroll
    for (int j = 0; j < 7; ++j) if (vals[j] > best) { best = vals[j]; bj = j; }
    float mx = best;
    for (int off = 16; off > 0; off >>= 1) mx = fmaxf(mx, __shfl_xor(mx, off, 32));
    int cand = (bj >= 0 && best == mx) ? (lane + bj * 32) : 0x7FFFFFFF;
    for (int off = 16; off > 0; off >>= 1) cand = min(cand, __shfl_xor(cand, off, 32));
    if (cand >= 0 && lane == (cand & 31) && (cand >> 5) < 7) vals[cand >> 5] = -INFINITY;
    if (lane == 0) topk[b * KSEL + k] = cand;
  }
}

// ---------- K5: WMMA dots over gathered top-K rows + fused loss reduction ----------
__global__ __launch_bounds__(128) void k5_wmma(const float* __restrict__ vit,
                                               const float* __restrict__ beta,
                                               const float* __restrict__ nE2,
                                               const float* __restrict__ nL2,
                                               const float* __restrict__ fE2,
                                               const int* __restrict__ topk,
                                               const unsigned short* __restrict__ LbH_u,
                                               const unsigned short* __restrict__ LbL_u,
                                               float* __restrict__ partial) {
  __shared__ __align__(32) bf16_t AsH[KSEL * ND];   // 24KB
  __shared__ __align__(32) bf16_t AsL[KSEL * ND];   // 24KB
  __shared__ float betas[NB];
  __shared__ float nLs[NB];
  __shared__ float nEg[KSEL];
  __shared__ float red[4];
  __shared__ float s_fEfL;

  const int b = blockIdx.x;
  const int tid = threadIdx.x;
  const int lane = tid & 31;
  const int wid = tid >> 5;
  const bf16_t* LbH = (const bf16_t*)LbH_u;
  const bf16_t* LbL = (const bf16_t*)LbL_u;

  // gather selected E rows into LDS, split fp32 -> bf16 hi/lo
  for (int r = 0; r < KSEL; ++r) {
    const int row = topk[b * KSEL + r];
    const float* src = vit + ((size_t)(b * NLAY + LYR) * SEQ + 1 + row) * ND;
    for (int i = tid; i < ND; i += 128) {
      const float v = src[i];
      const bf16_t h = f2bf(v);
      AsH[r * ND + i] = h;
      AsL[r * ND + i] = f2bf(v - bf2f(h));
    }
  }
  if (tid < KSEL) nEg[tid] = sqrtf(nE2[b * NTOK + topk[b * KSEL + tid]]);
  if (tid < NB) { betas[tid] = beta[b * NB + tid]; nLs[tid] = sqrtf(nL2[tid]); }
  if (tid == 0) {
    float s = 0.f;
    for (int m = 0; m < NB; ++m) s += nL2[m];
    s_fEfL = sqrtf(fE2[b]) * sqrtf(s);
  }
  __syncthreads();

  // per-wave 16x16 output tile: rows = gathered top-K, cols m0..m0+15
  const int half = lane >> 4;       // K-half selector
  const int lr   = lane & 15;
  const int m0   = wid * 16;
  const int aoff = half * 16;
  const bf16_t* ah = AsH + lr * ND + aoff;
  const bf16_t* al = AsL + lr * ND + aoff;
  const bf16_t* bh = LbH + (size_t)(m0 + lr) * ND + half * 16;
  const bf16_t* bl = LbL + (size_t)(m0 + lr) * ND + half * 16;

  v8f c = {};
  for (int k0 = 0; k0 < ND; k0 += 32) {
    union { v16bf v; struct { v8bf lo; v8bf hi; } p; } aH, aL;
    aH.p.lo = *(const v8bf*)(ah + k0);
    aH.p.hi = *(const v8bf*)(ah + k0 + 8);
    aL.p.lo = *(const v8bf*)(al + k0);
    aL.p.hi = *(const v8bf*)(al + k0 + 8);
    const v16bf bH = *(const v16bf*)(bh + k0);
    const v16bf bL = *(const v16bf*)(bl + k0);
    // compensated bf16 product: hi*hi + hi*lo + lo*hi  (~fp32 accuracy)
    c = __builtin_amdgcn_wmma_f32_16x16x32_bf16(false, aH.v, false, bH, (short)0, c, false, false);
    c = __builtin_amdgcn_wmma_f32_16x16x32_bf16(false, aH.v, false, bL, (short)0, c, false, false);
    c = __builtin_amdgcn_wmma_f32_16x16x32_bf16(false, aL.v, false, bH, (short)0, c, false, false);
  }

  const float fEfL = s_fEfL;
  const int m = m0 + lr;
  const float bm = betas[m];
  const float nl = nLs[m];
  float acc = 0.f;
#pragma unroll
  for (int r = 0; r < 8; ++r) {
    const int M = r + half * 8;                       // gathered-row index
    const float dv = c[r];
    const float cosv = dv / fmaxf(nEg[M] * nl, 1e-6f);
    acc += bm * expf(cosv) * (1.0f - dv / fEfL);
  }
  acc = wave_sum(acc);
  if (lane == 0) red[wid] = acc;
  __syncthreads();
  if (tid == 0) partial[b] = red[0] + red[1] + red[2] + red[3];
}

// ---------- K6: final mean (forward==backward => factor 2) ----------
__global__ __launch_bounds__(32) void k6_final(const float* __restrict__ partial,
                                               float* __restrict__ out) {
  if (threadIdx.x == 0) {
    float s = 0.f;
    for (int i = 0; i < NB; ++i) s += partial[i];
    out[0] = s * (2.0f / (float)NB);
  }
}

extern "C" void kernel_launch(void* const* d_in, const int* in_sizes, int n_in,
                              void* d_out, int out_size, void* d_ws, size_t ws_size,
                              hipStream_t stream) {
  (void)in_sizes; (void)n_in; (void)out_size; (void)ws_size;
  const float* vit    = (const float*)d_in[0];
  const float* bert   = (const float*)d_in[1];
  const float* labels = (const float*)d_in[2];
  float* out = (float*)d_out;

  float* ws      = (float*)d_ws;
  float* beta    = ws;                        // 64*64
  float* ny      = beta + NB * NB;            // 64*64
  float* o       = ny + NB * NB;              // 64*768
  float* nL2     = o + NB * ND;               // 64
  float* score   = nL2 + NB;                  // 64*196
  float* nE2     = score + NB * NTOK;         // 64*196
  float* fE2     = nE2 + NB * NTOK;           // 64
  float* fE2part = fE2 + NB;                  // 64*7
  float* partial = fE2part + NB * CHUNKS;     // 64
  int*   topk    = (int*)(partial + NB);      // 64*16
  unsigned short* LbH = (unsigned short*)(topk + NB * KSEL);  // 64*768 bf16
  unsigned short* LbL = LbH + NB * ND;                        // 64*768 bf16

  k1_softmax<<<1, 64, 0, stream>>>(labels, beta, ny);
  k2_nl<<<NB, 256, 0, stream>>>(bert, nL2);
  k2_o<<<(NB * ND) / 256, 256, 0, stream>>>(ny, bert, o);
  k2_split<<<(NB * ND) / 256, 256, 0, stream>>>(bert, LbH, LbL);
  k3_score<<<dim3(NB, CHUNKS), 256, 0, stream>>>(vit, o, score, nE2, fE2part);
  k3b_reduce<<<1, 64, 0, stream>>>(fE2part, fE2);
  k4_topk<<<NB, 32, 0, stream>>>(score, topk);
  k5_wmma<<<NB, 128, 0, stream>>>(vit, beta, nE2, nL2, fE2, topk, LbH, LbL, partial);
  k6_final<<<1, 32, 0, stream>>>(partial, out);
}
